// CPNKIFluxSingleAttention_54743653155409
// MI455X (gfx1250) — compile-verified, matrix operand order checked by reading, not run on hardware
//
#include <hip/hip_runtime.h>

// Problem constants
#define SQ 3072   // sequence length
#define DM 3072   // model dim (= H * DH)
#define HN 24     // heads
#define DH 128    // head dim

typedef __attribute__((ext_vector_type(16))) __bf16          v16bf;
typedef __attribute__((ext_vector_type(8)))  float           v8f;
typedef __attribute__((ext_vector_type(8)))  unsigned int    v8u;

// ---------------------------------------------------------------------------
// float -> bf16 (round to nearest even), stored as raw u16
// ---------------------------------------------------------------------------
__device__ __forceinline__ unsigned short f2bf(float f) {
  unsigned u = __builtin_bit_cast(unsigned, f);
  u += 0x7FFFu + ((u >> 16) & 1u);
  return (unsigned short)(u >> 16);
}

// ---------------------------------------------------------------------------
// WMMA 16x16x32 bf16 fragment loader (wave32).
// ISA layout (16-bit A-matrix 16x32): lane<16 holds M=lane with K
// {0..7,16..23}; lane>=16 holds M=lane-16 with K {8..15,24..31}.  B mirrors
// this with the lane indexing the column.  Half-pair j -> K offsets are
// {0,2,4,6,16,18,20,22} + (hi?8:0): consecutive-K u32 loads, which the
// backend merges into global_load_b128 / ds_load_b128.
// p must point at (row/col base) + kblock + (hi ? 8 : 0).
// ---------------------------------------------------------------------------
__device__ __forceinline__ v16bf load_frag_contig(const unsigned short* p) {
  v8u r;
#pragma unroll
  for (int jj = 0; jj < 8; ++jj) {
    const int off = (jj < 4) ? (2 * jj) : (16 + 2 * (jj - 4));
    r[jj] = *(const unsigned int*)(p + off);
  }
  return __builtin_bit_cast(v16bf, r);
}

__device__ __forceinline__ v8f wmma_bf16(v16bf a, v16bf b, v8f c) {
  // (neg_a, A, neg_b, B, c_mod, C, reuse_a, reuse_b)
  return __builtin_amdgcn_wmma_f32_16x16x32_bf16(false, a, false, b,
                                                 (short)0, c, false, false);
}

// ---------------------------------------------------------------------------
// Kernel 1: f32 -> bf16 conversion
// ---------------------------------------------------------------------------
__global__ __launch_bounds__(256)
void cvt_f32_bf16(const float* __restrict__ src,
                  unsigned short* __restrict__ dst, int n) {
  int i = blockIdx.x * blockDim.x + threadIdx.x;
  if (i < n) dst[i] = f2bf(src[i]);
}

// ---------------------------------------------------------------------------
// Kernel 2: C = A(M,K) * W(N,K)^T + bias   (bf16 in, f32 out)
// Register-blocked: one wave computes a 64x64 tile (4x4 WMMA sub-tiles),
// 16 WMMAs per 32-deep k-step from 4 A-frags + 4 B-frags (4x fragment
// reuse, 1 b128 load per WMMA).  Block = 256 threads = 8 waves arranged
// 2(M) x 4(N): workgroup tile 128x256.  grid = (M/128, N/256).
// ---------------------------------------------------------------------------
__global__ __launch_bounds__(256)
void gemm_xwT_bias(const unsigned short* __restrict__ A,
                   const unsigned short* __restrict__ W,
                   const float* __restrict__ bias,
                   float* __restrict__ C) {
  const int K = DM, N = DM;
  const int wave = threadIdx.x >> 5;
  const int lane = threadIdx.x & 31;
  const int hi = lane >> 4;
  const int ln = lane & 15;
  const int wm = wave >> 2;      // 0..1
  const int wn = wave & 3;       // 0..3
  const int rb = blockIdx.x * 128 + wm * 64;  // row base of 64x64 wave tile
  const int cb = blockIdx.y * 256 + wn * 64;  // col base

  const unsigned short* ap[4];
  const unsigned short* bp[4];
#pragma unroll
  for (int i = 0; i < 4; ++i) {
    ap[i] = A + (size_t)(rb + i * 16 + ln) * K + hi * 8;
    bp[i] = W + (size_t)(cb + i * 16 + ln) * K + hi * 8;
  }

  v8f acc[4][4];
#pragma unroll
  for (int i = 0; i < 4; ++i)
#pragma unroll
    for (int j = 0; j < 4; ++j) acc[i][j] = (v8f){};

  for (int kb = 0; kb < K; kb += 32) {
    v16bf am[4], bn[4];
#pragma unroll
    for (int i = 0; i < 4; ++i) {
      __builtin_prefetch(ap[i] + kb + 256, 0, 3);
      am[i] = load_frag_contig(ap[i] + kb);
    }
#pragma unroll
    for (int j = 0; j < 4; ++j) {
      __builtin_prefetch(bp[j] + kb + 256, 0, 3);
      bn[j] = load_frag_contig(bp[j] + kb);
    }
#pragma unroll
    for (int i = 0; i < 4; ++i)
#pragma unroll
      for (int j = 0; j < 4; ++j)
        acc[i][j] = wmma_bf16(am[i], bn[j], acc[i][j]);
  }

#pragma unroll
  for (int j = 0; j < 4; ++j) {
    const float bb = bias[cb + j * 16 + ln];
#pragma unroll
    for (int i = 0; i < 4; ++i) {
      float* cp = C + (size_t)(rb + i * 16 + hi * 8) * N + cb + j * 16 + ln;
#pragma unroll
      for (int r = 0; r < 8; ++r) cp[(size_t)r * N] = acc[i][j][r] + bb;
    }
  }
}

// ---------------------------------------------------------------------------
// Kernel 3: per-(s,h) RMSNorm + RoPE (mode 0, row-major bf16 out) or plain
// convert with per-head transpose for V (mode 1: out[(h*DH+d)*SQ + s]).
// One wave per (s,h); each lane owns 4 consecutive head-dim elements
// (two RoPE pairs).  grid = (S, H), block = 32.
// ---------------------------------------------------------------------------
__global__ __launch_bounds__(32)
void norm_rope_cvt(const float* __restrict__ X, const float* __restrict__ g,
                   const float* __restrict__ cosb, const float* __restrict__ sinb,
                   unsigned short* __restrict__ out, int mode) {
  const int s = blockIdx.x, h = blockIdx.y, lane = threadIdx.x;
  const size_t base = ((size_t)s * HN + h) * DH;
  const int d = lane * 4;

  const float4 xv = *(const float4*)(X + base + d);
  float x0 = xv.x, x1 = xv.y, x2 = xv.z, x3 = xv.w;

  if (mode == 0) {
    float ss = x0 * x0 + x1 * x1 + x2 * x2 + x3 * x3;
#pragma unroll
    for (int msk = 1; msk <= 16; msk <<= 1) ss += __shfl_xor(ss, msk, 32);
    const float inv = rsqrtf(ss * (1.0f / DH) + 1e-6f);
    x0 *= inv * g[d + 0];
    x1 *= inv * g[d + 1];
    x2 *= inv * g[d + 2];
    x3 *= inv * g[d + 3];
    const float4 cv = *(const float4*)(cosb + (size_t)s * DH + d);
    const float4 sv = *(const float4*)(sinb + (size_t)s * DH + d);
    const float r0 = x0 * cv.x - x1 * sv.x;  // even: x*cos - x_odd*sin
    const float r1 = x1 * cv.y + x0 * sv.y;  // odd : x*cos + x_even*sin
    const float r2 = x2 * cv.z - x3 * sv.z;
    const float r3 = x3 * cv.w + x2 * sv.w;
    out[base + d + 0] = f2bf(r0);
    out[base + d + 1] = f2bf(r1);
    out[base + d + 2] = f2bf(r2);
    out[base + d + 3] = f2bf(r3);
  } else {
    // V path: convert + transpose to Vt[h][d][s] so the P*V B-fragments
    // become contiguous b128 loads in the attention kernel.
    unsigned short* vt = out + ((size_t)h * DH + d) * SQ + s;
    vt[0 * (size_t)SQ] = f2bf(x0);
    vt[1 * (size_t)SQ] = f2bf(x1);
    vt[2 * (size_t)SQ] = f2bf(x2);
    vt[3 * (size_t)SQ] = f2bf(x3);
  }
}

// ---------------------------------------------------------------------------
// Kernel 4: flash attention.  One wave per (16-query tile, head).
// 64-key steps: Q fragments reused across 4 score tiles; online softmax on
// C-fragments; P (16x64) goes through a 2 KB LDS tile to convert C-layout
// into A-layout; P*V reads pre-transposed Vt with contiguous fragments.
// grid = (S/16, H), block = 32.
// ---------------------------------------------------------------------------
__global__ __launch_bounds__(32)
void flash_attn(const unsigned short* __restrict__ Q,
                const unsigned short* __restrict__ Km,
                const unsigned short* __restrict__ Vt,
                float* __restrict__ O) {
  __shared__ unsigned short pbuf[16 * 64];

  const int lane = threadIdx.x;
  const int hi = lane >> 4;
  const int ln = lane & 15;
  const int qt = blockIdx.x;
  const int h  = blockIdx.y;
  const float scale = 0.08838834764831845f;  // 1/sqrt(128)

  // Q fragments for this 16x128 block (A-matrix), loaded once.
  const unsigned short* qp = Q + (size_t)(qt * 16 + ln) * DM + h * DH + hi * 8;
  v16bf aq[4];
#pragma unroll
  for (int c = 0; c < 4; ++c) aq[c] = load_frag_contig(qp + c * 32);

  v8f acc[8];
#pragma unroll
  for (int t = 0; t < 8; ++t) acc[t] = (v8f){};
  float mrow[8], lrow[8];
#pragma unroll
  for (int r = 0; r < 8; ++r) { mrow[r] = -3.0e38f; lrow[r] = 0.0f; }

  for (int kb = 0; kb < SQ; kb += 64) {
    // ---- scores S = Q * K^T for 64 keys (four 16-key C tiles) ----
    v8f sc[4];
#pragma unroll
    for (int t = 0; t < 4; ++t) {
      const unsigned short* kp =
          Km + (size_t)(kb + t * 16 + ln) * DM + h * DH + hi * 8;
      v8f s = (v8f){};
#pragma unroll
      for (int c = 0; c < 4; ++c)
        s = wmma_bf16(aq[c], load_frag_contig(kp + c * 32), s);
      sc[t] = s;
    }

    // ---- online softmax (row stats replicated across each 16-lane half) ----
    float p[4][8];
#pragma unroll
    for (int r = 0; r < 8; ++r) {
      float v[4];
      float mx = -3.0e38f;
#pragma unroll
      for (int t = 0; t < 4; ++t) {
        v[t] = sc[t][r] * scale;
        mx = fmaxf(mx, v[t]);
      }
#pragma unroll
      for (int msk = 1; msk <= 8; msk <<= 1)
        mx = fmaxf(mx, __shfl_xor(mx, msk, 32));
      const float mnew = fmaxf(mrow[r], mx);
      float rs = 0.0f;
#pragma unroll
      for (int t = 0; t < 4; ++t) {
        p[t][r] = __expf(v[t] - mnew);
        rs += p[t][r];
      }
#pragma unroll
      for (int msk = 1; msk <= 8; msk <<= 1) rs += __shfl_xor(rs, msk, 32);
      const float corr = __expf(mrow[r] - mnew);
      lrow[r] = lrow[r] * corr + rs;
      mrow[r] = mnew;
#pragma unroll
      for (int t = 0; t < 8; ++t) acc[t][r] *= corr;
    }

    // ---- P: C-layout -> LDS 16x64 bf16 tile -> A-layout fragments ----
#pragma unroll
    for (int r = 0; r < 8; ++r) {
      const int row = r + hi * 8;
#pragma unroll
      for (int t = 0; t < 4; ++t)
        pbuf[row * 64 + t * 16 + ln] = f2bf(p[t][r]);
    }
    __syncthreads();
    v16bf pa[2];
#pragma unroll
    for (int c = 0; c < 2; ++c)
      pa[c] = load_frag_contig(&pbuf[ln * 64 + c * 32 + hi * 8]);
    __syncthreads();

    // ---- O += P * V  (8 output column tiles, 2 k-chunks of 32 keys) ----
#pragma unroll
    for (int t = 0; t < 8; ++t) {
      const unsigned short* vp =
          Vt + ((size_t)h * DH + t * 16 + ln) * SQ + kb + hi * 8;
#pragma unroll
      for (int c = 0; c < 2; ++c)
        acc[t] = wmma_bf16(pa[c], load_frag_contig(vp + c * 32), acc[t]);
    }
  }

  // ---- epilogue: normalize and store f32 output ----
#pragma unroll
  for (int r = 0; r < 8; ++r) {
    const float inv = 1.0f / lrow[r];
    const size_t row = (size_t)(qt * 16 + r + hi * 8);
#pragma unroll
    for (int t = 0; t < 8; ++t)
      O[row * DM + h * DH + t * 16 + ln] = acc[t][r] * inv;
  }
}

// ---------------------------------------------------------------------------
// Host-side launch
// ---------------------------------------------------------------------------
extern "C" void kernel_launch(void* const* d_in, const int* in_sizes, int n_in,
                              void* d_out, int out_size, void* d_ws, size_t ws_size,
                              hipStream_t stream) {
  const float* hidden = (const float*)d_in[0];
  const float* cosb   = (const float*)d_in[1];
  const float* sinb   = (const float*)d_in[2];
  const float* wq     = (const float*)d_in[3];
  const float* bq     = (const float*)d_in[4];
  const float* wk     = (const float*)d_in[5];
  const float* bk     = (const float*)d_in[6];
  const float* wv     = (const float*)d_in[7];
  const float* bv     = (const float*)d_in[8];
  const float* gq     = (const float*)d_in[9];
  const float* gk     = (const float*)d_in[10];
  float* out = (float*)d_out;

  const size_t SZ = (size_t)SQ * DM;  // 9,437,184 elements

  // Workspace layout
  unsigned short* Xb  = (unsigned short*)d_ws;
  unsigned short* Wqb = Xb + SZ;
  unsigned short* Wkb = Wqb + SZ;
  unsigned short* Wvb = Wkb + SZ;
  float* Qf = (float*)(Wvb + SZ);
  float* Kf = Qf + SZ;
  float* Vf = Kf + SZ;
  unsigned short* Qb  = (unsigned short*)(Vf + SZ);
  unsigned short* Kb  = Qb + SZ;
  unsigned short* Vtb = Kb + SZ;   // per-head transposed V

  const int n = (int)SZ;
  const int cvtBlocks = (n + 255) / 256;
  cvt_f32_bf16<<<cvtBlocks, 256, 0, stream>>>(hidden, Xb, n);
  cvt_f32_bf16<<<cvtBlocks, 256, 0, stream>>>(wq, Wqb, n);
  cvt_f32_bf16<<<cvtBlocks, 256, 0, stream>>>(wk, Wkb, n);
  cvt_f32_bf16<<<cvtBlocks, 256, 0, stream>>>(wv, Wvb, n);

  dim3 gg(SQ / 128, DM / 256);
  gemm_xwT_bias<<<gg, 256, 0, stream>>>(Xb, Wqb, bq, Qf);
  gemm_xwT_bias<<<gg, 256, 0, stream>>>(Xb, Wkb, bk, Kf);
  gemm_xwT_bias<<<gg, 256, 0, stream>>>(Xb, Wvb, bv, Vf);

  dim3 gn(SQ, HN);
  norm_rope_cvt<<<gn, 32, 0, stream>>>(Qf, gq, cosb, sinb, Qb, 0);
  norm_rope_cvt<<<gn, 32, 0, stream>>>(Kf, gk, cosb, sinb, Kb, 0);
  norm_rope_cvt<<<gn, 32, 0, stream>>>(Vf, nullptr, nullptr, nullptr, Vtb, 1);

  dim3 ga(SQ / 16, HN);
  flash_attn<<<ga, 32, 0, stream>>>(Qb, Kb, Vtb, out);
}
